// LightGATLayer_13967233646640
// MI455X (gfx1250) — matrix operand order repaired and checked by reading, not compiled, not run on hardware
//
#include <hip/hip_runtime.h>
#include <hip/hip_bf16.h>

#define DIM 128

typedef __attribute__((ext_vector_type(16))) _Float16 v16h;
typedef __attribute__((ext_vector_type(8)))  float    v8f;

// Monotone order-preserving f32 <-> u32 mapping (for atomicMax on floats incl. negatives)
__device__ __forceinline__ unsigned int f32_ordered(float x) {
  unsigned int u = __float_as_uint(x);
  return (u & 0x80000000u) ? ~u : (u | 0x80000000u);
}
__device__ __forceinline__ float ordered_f32(unsigned int u) {
  return (u & 0x80000000u) ? __uint_as_float(u & 0x7FFFFFFFu) : __uint_as_float(~u);
}

// ---------------------------------------------------------------- init
__global__ void init_node_state(unsigned int* __restrict__ umaxb,
                                unsigned int* __restrict__ imaxb,
                                float* __restrict__ usum,
                                float* __restrict__ isum,
                                int nu, int ni) {
  int t = blockIdx.x * blockDim.x + threadIdx.x;
  const unsigned int neg_inf_key = 0x007FFFFFu; // f32_ordered(-inf)
  if (t < nu) { umaxb[t] = neg_inf_key; usum[t] = 0.0f; }
  if (t < ni) { imaxb[t] = neg_inf_key; isum[t] = 0.0f; }
}

// ---------------------------------------------------------------- phase 1: edge logits via WMMA
// One wave32 computes logits for 16 edges.
// A (16x32 f16): row m = gathered embedding chunk of edge (ebase+m); lane = m + 16*half,
//   per-lane halves: K = half*8 + {0..7} and K = 16 + half*8 + {0..7}   (ISA 16-bit A layout)
// B (32x16 f16): column n = attn_w chunk, replicated across all 16 columns; per-lane halves:
//   K = half*8 + {0..7} and K = 16 + half*8 + {0..7} for column (lane%16)  -> every C column equals the dot product.
// C (16x16 f32): row M=m lives in c[m%8] of lanes {0..15 : m<8, 16..31 : m>=8}; all columns equal,
//   so lane 0 holds raw[0..7] and lane 16 holds raw[8..15].
__global__ __launch_bounds__(256)
void edge_scores_wmma(const float* __restrict__ u_emb,
                      const float* __restrict__ i_emb,
                      const int* __restrict__ user_idx,
                      const int* __restrict__ item_idx,
                      const float* __restrict__ attn_w,
                      float* __restrict__ raw_out,
                      int num_edges) {
  const int lane  = threadIdx.x & 31;
  const int wave  = threadIdx.x >> 5;
  const int ebase = (blockIdx.x * 8 + wave) * 16;
  if (ebase >= num_edges) return;          // wave-uniform exit, EXEC stays all-ones for live waves

  const int m    = lane & 15;
  const int half = lane >> 4;

  int e = ebase + m;
  if (e >= num_edges) e = num_edges - 1;   // clamp (keeps EXEC full through WMMA); tail stores guarded
  const float* __restrict__ urow = u_emb + (long long)user_idx[e] * DIM;
  const float* __restrict__ irow = i_emb + (long long)item_idx[e] * DIM;

  v8f c = {};
  #pragma unroll
  for (int ch = 0; ch < 8; ++ch) {         // K = 256 total: 0..127 -> u*w_u, 128..255 -> i*w_i
    const int kbase = ch * 32;
    const float* __restrict__ row = (kbase < DIM) ? urow : irow;
    const int koff = kbase & (DIM - 1);

    const float* pa0 = row + koff + half * 8;
    const float* pa1 = row + koff + 16 + half * 8;
    float4 f0 = *(const float4*)(pa0);
    float4 f1 = *(const float4*)(pa0 + 4);
    float4 f2 = *(const float4*)(pa1);
    float4 f3 = *(const float4*)(pa1 + 4);
    v16h a;
    a[0]=(_Float16)f0.x;  a[1]=(_Float16)f0.y;  a[2]=(_Float16)f0.z;  a[3]=(_Float16)f0.w;
    a[4]=(_Float16)f1.x;  a[5]=(_Float16)f1.y;  a[6]=(_Float16)f1.z;  a[7]=(_Float16)f1.w;
    a[8]=(_Float16)f2.x;  a[9]=(_Float16)f2.y;  a[10]=(_Float16)f2.z; a[11]=(_Float16)f2.w;
    a[12]=(_Float16)f3.x; a[13]=(_Float16)f3.y; a[14]=(_Float16)f3.z; a[15]=(_Float16)f3.w;

    const float* pw = attn_w + kbase + half * 8;
    float4 w0 = *(const float4*)(pw);
    float4 w1 = *(const float4*)(pw + 4);
    float4 w2 = *(const float4*)(pw + 16);
    float4 w3 = *(const float4*)(pw + 20);
    v16h b;
    b[0]=(_Float16)w0.x;  b[1]=(_Float16)w0.y;  b[2]=(_Float16)w0.z;  b[3]=(_Float16)w0.w;
    b[4]=(_Float16)w1.x;  b[5]=(_Float16)w1.y;  b[6]=(_Float16)w1.z;  b[7]=(_Float16)w1.w;
    b[8]=(_Float16)w2.x;  b[9]=(_Float16)w2.y;  b[10]=(_Float16)w2.z; b[11]=(_Float16)w2.w;
    b[12]=(_Float16)w3.x; b[13]=(_Float16)w3.y; b[14]=(_Float16)w3.z; b[15]=(_Float16)w3.w;

    c = __builtin_amdgcn_wmma_f32_16x16x32_f16(
        /*neg_a=*/false, a, /*neg_b=*/false, b,
        /*c_mod=*/(short)0, c, /*reuse_a=*/false, /*reuse_b=*/false);
  }

  if (m == 0) {                            // lanes 0 and 16 hold rows half*8 .. half*8+7
    #pragma unroll
    for (int j = 0; j < 8; ++j) {
      int eo = ebase + half * 8 + j;
      if (eo < num_edges) {
        float x = c[j];
        raw_out[eo] = (x >= 0.0f) ? x : 0.2f * x;   // LeakyReLU(0.2)
      }
    }
  }
}

// ---------------------------------------------------------------- phase 2: segment max
__global__ void edge_seg_max(const float* __restrict__ raw,
                             const int* __restrict__ user_idx,
                             const int* __restrict__ item_idx,
                             unsigned int* __restrict__ umaxb,
                             unsigned int* __restrict__ imaxb,
                             int num_edges) {
  int e = blockIdx.x * blockDim.x + threadIdx.x;
  if (e >= num_edges) return;
  unsigned int key = f32_ordered(raw[e]);
  atomicMax(&umaxb[user_idx[e]], key);
  atomicMax(&imaxb[item_idx[e]], key);
}

// ---------------------------------------------------------------- phase 3: exp + segment sum
__global__ void edge_exp_sum(const float* __restrict__ raw,
                             const int* __restrict__ user_idx,
                             const int* __restrict__ item_idx,
                             const unsigned int* __restrict__ umaxb,
                             const unsigned int* __restrict__ imaxb,
                             float* __restrict__ eu, float* __restrict__ ei,
                             float* __restrict__ usum, float* __restrict__ isum,
                             int num_edges) {
  int e = blockIdx.x * blockDim.x + threadIdx.x;
  if (e >= num_edges) return;
  float r = raw[e];
  int u = user_idx[e], it = item_idx[e];
  float xu = __expf(r - ordered_f32(umaxb[u]));
  float xi = __expf(r - ordered_f32(imaxb[it]));
  eu[e] = xu;
  ei[e] = xi;
  atomicAdd(&usum[u], xu);
  atomicAdd(&isum[it], xi);
}

// ---------------------------------------------------------------- phase 4: attention-weighted scatter-add
// 256 threads = 2 edges x 128 dims per block.
__global__ __launch_bounds__(256)
void edge_scatter(const float* __restrict__ u_emb,
                  const float* __restrict__ i_emb,
                  const int* __restrict__ user_idx,
                  const int* __restrict__ item_idx,
                  const float* __restrict__ eu, const float* __restrict__ ei,
                  const float* __restrict__ usum, const float* __restrict__ isum,
                  float* __restrict__ new_u, float* __restrict__ new_i,
                  int num_edges) {
  int e = blockIdx.x * 2 + (threadIdx.x >> 7);
  int d = threadIdx.x & (DIM - 1);
  if (e >= num_edges) return;
  int u = user_idx[e], it = item_idx[e];
  float ua = eu[e] / (usum[u]  + 1e-10f);
  float ia = ei[e] / (isum[it] + 1e-10f);
  atomicAdd(&new_u[(long long)u  * DIM + d], i_emb[(long long)it * DIM + d] * ua);
  atomicAdd(&new_i[(long long)it * DIM + d], u_emb[(long long)u  * DIM + d] * ia);
}

// ---------------------------------------------------------------- launcher
extern "C" void kernel_launch(void* const* d_in, const int* in_sizes, int n_in,
                              void* d_out, int out_size, void* d_ws, size_t ws_size,
                              hipStream_t stream) {
  const float* u_emb  = (const float*)d_in[0];
  const float* i_emb  = (const float*)d_in[1];
  const int*   edges  = (const int*)d_in[2];
  // d_in[3] = weights: unused by the reference computation
  const float* attn_w = (const float*)d_in[4];

  const int NU = in_sizes[0] / DIM;
  const int NI = in_sizes[1] / DIM;
  const int NE = in_sizes[2] / 2;
  const int* user_idx = edges;
  const int* item_idx = edges + NE;

  // workspace layout (all 4-byte elements)
  float*        raw   = (float*)d_ws;
  float*        eu    = raw + NE;
  float*        ei    = eu + NE;
  unsigned int* umaxb = (unsigned int*)(ei + NE);
  unsigned int* imaxb = umaxb + NU;
  float*        usum  = (float*)(imaxb + NI);
  float*        isum  = usum + NU;

  float* new_u = (float*)d_out;
  float* new_i = new_u + (long long)NU * DIM;

  // output starts as a copy of the embeddings (.at[].add semantics)
  hipMemcpyAsync(new_u, u_emb, (size_t)NU * DIM * sizeof(float),
                 hipMemcpyDeviceToDevice, stream);
  hipMemcpyAsync(new_i, i_emb, (size_t)NI * DIM * sizeof(float),
                 hipMemcpyDeviceToDevice, stream);

  int nmax = (NU > NI) ? NU : NI;
  init_node_state<<<(nmax + 255) / 256, 256, 0, stream>>>(umaxb, imaxb, usum, isum, NU, NI);

  int ntiles = (NE + 15) / 16;
  edge_scores_wmma<<<(ntiles + 7) / 8, 256, 0, stream>>>(
      u_emb, i_emb, user_idx, item_idx, attn_w, raw, NE);

  edge_seg_max<<<(NE + 255) / 256, 256, 0, stream>>>(
      raw, user_idx, item_idx, umaxb, imaxb, NE);

  edge_exp_sum<<<(NE + 255) / 256, 256, 0, stream>>>(
      raw, user_idx, item_idx, umaxb, imaxb, eu, ei, usum, isum, NE);

  edge_scatter<<<(NE + 1) / 2, 256, 0, stream>>>(
      u_emb, i_emb, user_idx, item_idx, eu, ei, usum, isum, new_u, new_i, NE);
}
